// RND1DecoderLayer_11046655885390
// MI455X (gfx1250) — matrix operand order repaired and checked by reading, not compiled, not run on hardware
//
#include <hip/hip_runtime.h>
#include <hip/hip_bf16.h>
#include <stdint.h>

// ---------- model dims ----------
#define DH    2048   // hidden
#define NHEAD 16
#define NKV   4
#define HD    128
#define NEXP  16
#define TOPK  2
#define IMOE  768
#define SEQ   1024
#define NPAIR (SEQ*TOPK)
#define CAP   SEQ     // per-expert capacity (worst case)

typedef unsigned short u16;
typedef unsigned int   u32;
typedef __bf16 bf16;
typedef bf16  v16bf __attribute__((ext_vector_type(16)));
typedef float v8f   __attribute__((ext_vector_type(8)));
typedef float v4f   __attribute__((ext_vector_type(4)));
typedef u32   v2u   __attribute__((ext_vector_type(2)));
typedef u32   v4u   __attribute__((ext_vector_type(4)));
typedef int   v8i   __attribute__((ext_vector_type(8)));
typedef int   v4i   __attribute__((ext_vector_type(4)));

union Frag { v16bf bf; v4f f4[2]; };

__device__ __forceinline__ u16 f2bf(float f){
  u32 u = __builtin_bit_cast(u32, f);
  u += 0x7fffu + ((u >> 16) & 1u);           // round-to-nearest-even
  return (u16)(u >> 16);
}
__device__ __forceinline__ float bf2f(u16 h){
  return __builtin_bit_cast(float, (u32)h << 16);
}
__device__ __forceinline__ v8f vzero8(){ v8f z = {0.f,0.f,0.f,0.f,0.f,0.f,0.f,0.f}; return z; }

// ---------------------------------------------------------------------------
// TDM: DMA a dense 2D bf16 tile [rows x cols] (row stride = strideElems) from
// global memory into LDS (packed row-major). D# layout per CDNA5 ISA ch.8.
// Issued once per wave; completion via TENSORcnt.
// clang-23 toolchain: 6-arg builtin (g0 u32x4, g1 i32x8, i32x4, i32x4, i32x8, cpol)
// ---------------------------------------------------------------------------
__device__ __forceinline__ void tdm_load_tile_bf16(
    const u16* gsrc, void* lds_dst, int rows, int cols, int strideElems)
{
  u32 ldsAddr = (u32)(uintptr_t)lds_dst;                 // LDS flat addr low 32b
  unsigned long long ga = (unsigned long long)(uintptr_t)gsrc;
  v4u g0;
  g0[0] = 1u;                                            // count=1, user, no gather
  g0[1] = ldsAddr;                                       // lds_addr
  g0[2] = (u32)(ga & 0xffffffffu);                       // global_addr[31:0]
  g0[3] = (u32)((ga >> 32) & 0x1ffffffu) | (2u << 30);   // global_addr[56:32], type=2
  v8i g1;
  g1[0] = (int)(1u << 16);                               // data_size=1 (2 bytes)
  g1[1] = (int)((u32)(cols & 0xffff) << 16);             // tensor_dim0[15:0]
  g1[2] = (int)(((u32)(cols >> 16) & 0xffff) |           // tensor_dim0[31:16]
                ((u32)(rows & 0xffff) << 16));           // tensor_dim1[15:0]
  g1[3] = (int)(((u32)(rows >> 16) & 0xffff) |           // tensor_dim1[31:16]
                ((u32)(cols & 0xffff) << 16));           // tile_dim0 = cols
  g1[4] = (int)(rows & 0xffff);                          // tile_dim1 = rows, tile_dim2=0
  g1[5] = (int)strideElems;                              // tensor_dim0_stride[31:0]
  g1[6] = 0;                                             // stride[47:32], dim1_stride lo
  g1[7] = 0;
  v4i g2 = {0,0,0,0};
  v4i g3 = {0,0,0,0};
  v8i g4 = {0,0,0,0,0,0,0,0};
  __builtin_amdgcn_tensor_load_to_lds(g0, g1, g2, g3, g4, 0);
}

// ---------------------------------------------------------------------------
// RMSNorm (fp32 in) -> bf16 out
// ---------------------------------------------------------------------------
__global__ __launch_bounds__(256) void k_rmsnorm_bf16(
    const float* __restrict__ x, const float* __restrict__ w,
    u16* __restrict__ y, int D)
{
  int row = blockIdx.x, tid = threadIdx.x, wid = tid >> 5, lane = tid & 31;
  const float* xr = x + (size_t)row * D;
  float ss = 0.f;
  for (int i = tid; i < D; i += 256){ float v = xr[i]; ss += v * v; }
  for (int d = 16; d; d >>= 1) ss += __shfl_xor(ss, d, 32);
  __shared__ float red[9];
  if (lane == 0) red[wid] = ss;
  __syncthreads();
  if (wid == 0){
    float t = (lane < 8) ? red[lane] : 0.f;
    for (int d = 4; d; d >>= 1) t += __shfl_xor(t, d, 32);
    if (lane == 0) red[8] = t;
  }
  __syncthreads();
  float r = rsqrtf(red[8] / (float)D + 1e-6f);
  u16* yr = y + (size_t)row * D;
  for (int i = tid; i < D; i += 256) yr[i] = f2bf(xr[i] * r * w[i]);
}

// ---------------------------------------------------------------------------
// Generic WMMA GEMM:  C[outrow, 0..N) = gatherA(bf16)[arow, 0..K) x W[N,K]^T
// W is fp32 row-major [nmats][N][K] (converted to bf16 into LDS).
// Block tile 128x128, BK=32, 256 threads = 8 waves (4x2), wave tile 32x64.
// Dense (no-gather) A tiles are staged via the Tensor Data Mover; gathered
// (MoE) A tiles are staged manually per row.
// ---------------------------------------------------------------------------
#define BM 128
#define BN 128
#define BK 32

__global__ __launch_bounds__(256) void k_gemm_bf16(
    const u16*  __restrict__ A, int K,
    const float* __restrict__ W, int N,
    float* __restrict__ C, int ldc,
    const float* __restrict__ residual,   // optional, [outrow][ldc]
    const float* __restrict__ rowScale,   // optional, per outrow
    const int*  __restrict__ rowmap,      // optional, [nmats][cap]
    const int*  __restrict__ counts,      // optional, [nmats]
    int Mcap, int cap, int aRowShift)     // arow = map >> aRowShift
{
  const int mat = blockIdx.z;
  const int m0  = blockIdx.y * BM;
  const int n0  = blockIdx.x * BN;
  const int count = counts ? counts[mat] : Mcap;
  if (m0 >= count) return;
  const float* Wm = W + (size_t)mat * N * K;

  __shared__ u16 lA[BM][BK];
  __shared__ u16 lB[BN][BK];
  __shared__ int rmapS[BM];
  __shared__ int arowS[BM];

  const int tid = threadIdx.x;
  if (tid < BM){
    int gi = m0 + tid, outr = -1, ar = -1;
    if (gi < count){
      if (rowmap){ int p = rowmap[mat * cap + gi]; outr = p; ar = p >> aRowShift; }
      else       { outr = gi; ar = gi; }
    }
    rmapS[tid] = outr; arowS[tid] = ar;
  }
  __syncthreads();

  const int wid = tid >> 5, lane = tid & 31;
  const int wm = wid & 3, wn = wid >> 2;       // 4 x 2 wave grid
  const int rb = wm * 32, cb = wn * 64;
  const int half = lane >> 4, l15 = lane & 15;
  const int kb0A = half ? 8 : 0;
  const int kb0B = half ? 8 : 0;

  v8f acc[2][4];
  #pragma unroll
  for (int i = 0; i < 2; ++i)
    #pragma unroll
    for (int j = 0; j < 4; ++j) acc[i][j] = vzero8();

  for (int k0 = 0; k0 < K; k0 += BK){
    // --- stage A (bf16) ---
    if (rowmap){
      #pragma unroll
      for (int it = 0; it < 2; ++it){
        int chunk = tid + it * 256;             // 0..511
        int row = chunk >> 2, kc = (chunk & 3) * 8;
        int ar = arowS[row];
        v4f z = {0.f,0.f,0.f,0.f};
        if (ar >= 0) z = *(const v4f*)(A + (size_t)ar * K + k0 + kc);
        *(v4f*)&lA[row][kc] = z;
      }
    } else if (wid == 0){
      tdm_load_tile_bf16(A + (size_t)m0 * K + k0, &lA[0][0], BM, BK, K);
    }
    // --- stage B (fp32 -> bf16) ---
    #pragma unroll
    for (int it = 0; it < 4; ++it){
      int chunk = tid + it * 256;               // 0..1023
      int br = chunk >> 3, bk = (chunk & 7) * 4;
      const float* src = Wm + (size_t)(n0 + br) * K + k0 + bk;
      v4f wv = *(const v4f*)src;
      if (it == 0 && k0 + BK < K) __builtin_prefetch(src + BK, 0, 0);
      u32 p0 = (u32)f2bf(wv[0]) | ((u32)f2bf(wv[1]) << 16);
      u32 p1 = (u32)f2bf(wv[2]) | ((u32)f2bf(wv[3]) << 16);
      v2u pk = {p0, p1};
      *(v2u*)&lB[br][bk] = pk;
    }
    if (!rowmap && wid == 0) __builtin_amdgcn_s_wait_tensorcnt(0);
    __syncthreads();

    Frag a[2], b[4];
    #pragma unroll
    for (int i = 0; i < 2; ++i){
      const u16* r = &lA[rb + i * 16 + l15][0];
      a[i].f4[0] = *(const v4f*)(r + kb0A);
      a[i].f4[1] = *(const v4f*)(r + kb0A + 16);
    }
    #pragma unroll
    for (int j = 0; j < 4; ++j){
      const u16* r = &lB[cb + j * 16 + l15][0];
      b[j].f4[0] = *(const v4f*)(r + kb0B);
      b[j].f4[1] = *(const v4f*)(r + kb0B + 16);
    }
    #pragma unroll
    for (int i = 0; i < 2; ++i)
      #pragma unroll
      for (int j = 0; j < 4; ++j)
        acc[i][j] = __builtin_amdgcn_wmma_f32_16x16x32_bf16(
            false, a[i].bf, false, b[j].bf, (short)0, acc[i][j], false, false);
    __syncthreads();
  }

  // --- epilogue: C layout row = v + 8*half, col = l15 ---
  #pragma unroll
  for (int i = 0; i < 2; ++i)
    #pragma unroll
    for (int j = 0; j < 4; ++j)
      #pragma unroll
      for (int v = 0; v < 8; ++v){
        int ti = rb + i * 16 + v + half * 8;
        int outr = rmapS[ti];
        if (outr < 0) continue;
        int col = n0 + cb + j * 16 + l15;
        float val = acc[i][j][v];
        if (rowScale) val *= rowScale[outr];
        if (residual) val += residual[(size_t)outr * ldc + col];
        C[(size_t)outr * ldc + col] = val;
      }
}

// ---------------------------------------------------------------------------
// Per-head RMSNorm (q/k) + RoPE -> bf16 [head][S][HD]; V -> bf16 transposed
// [KV][HD][S]. One block per token, 8 waves loop over 16q + 4k + 4v heads.
// ---------------------------------------------------------------------------
__global__ __launch_bounds__(256) void k_qk_rope_v(
    const float* __restrict__ qf, const float* __restrict__ kf,
    const float* __restrict__ vf,
    const float* __restrict__ qnw, const float* __restrict__ knw,
    const int* __restrict__ pos_ids,
    u16* __restrict__ Qb, u16* __restrict__ Kb, u16* __restrict__ VTb)
{
  const int s = blockIdx.x, tid = threadIdx.x, wid = tid >> 5, lane = tid & 31;
  const float pos = (float)pos_ids[s];
  for (int idx = wid; idx < 24; idx += 8){
    if (idx < 20){
      const bool isq = idx < 16;
      const int hh = isq ? idx : idx - 16;
      const float* src = isq ? qf + ((size_t)s * (NHEAD*HD) + hh * HD)
                             : kf + ((size_t)s * (NKV*HD)  + hh * HD);
      const float* wv = isq ? qnw : knw;
      const int d0 = lane * 4;
      float x[4]; float ss = 0.f;
      #pragma unroll
      for (int j = 0; j < 4; ++j){ x[j] = src[d0 + j]; ss += x[j] * x[j]; }
      for (int d = 16; d; d >>= 1) ss += __shfl_xor(ss, d, 32);
      float r = rsqrtf(ss * (1.f / HD) + 1e-6f);
      float n[4], o[4];
      #pragma unroll
      for (int j = 0; j < 4; ++j) n[j] = x[j] * r * wv[d0 + j];
      #pragma unroll
      for (int j = 0; j < 4; ++j){
        int d = d0 + j, fi = d & 63;
        float ang = pos * __powf(1.0e6f, -(float)fi * (1.f / 64.f));
        float partner = __shfl_xor(n[j], 16, 32);    // d ^ 64
        float rot = (d < 64) ? -partner : partner;
        o[j] = n[j] * __cosf(ang) + rot * __sinf(ang);
      }
      u16* dst = isq ? Qb + ((size_t)hh * SEQ + s) * HD
                     : Kb + ((size_t)hh * SEQ + s) * HD;
      #pragma unroll
      for (int j = 0; j < 4; ++j) dst[d0 + j] = f2bf(o[j]);
    } else {
      const int hh = idx - 20, d0 = lane * 4;
      #pragma unroll
      for (int j = 0; j < 4; ++j)
        VTb[((size_t)hh * HD + d0 + j) * SEQ + s] =
            f2bf(vf[(size_t)s * (NKV*HD) + hh * HD + d0 + j]);
    }
  }
}

// ---------------------------------------------------------------------------
// Flash-style full (non-causal) attention. One wave per (head, 16-query tile).
// QK^T and PV via v_wmma_f32_16x16x32_bf16; online softmax; P through LDS
// to convert C-layout -> A-layout.
// ---------------------------------------------------------------------------
__global__ __launch_bounds__(32) void k_attention(
    const u16* __restrict__ Q,   // [H][S][HD]
    const u16* __restrict__ Kc,  // [KV][S][HD]
    const u16* __restrict__ VT,  // [KV][HD][S]
    u16* __restrict__ O)         // [S][H*HD]
{
  const int h = blockIdx.y, kvh = h >> 2;
  const int q0 = blockIdx.x * 16;
  const int lane = threadIdx.x, half = lane >> 4, l15 = lane & 15;
  const int kb0A = half ? 8 : 0;      // A-frag K offsets
  const int kb0B = half ? 8 : 0;      // B-frag K offsets
  const float scale = 0.08838834764831845f;   // 1/sqrt(128)

  Frag qa[4];
  const u16* qbase = Q + ((size_t)h * SEQ + q0) * HD;
  #pragma unroll
  for (int hb = 0; hb < 4; ++hb){
    const u16* r = qbase + (size_t)l15 * HD + hb * 32;
    qa[hb].f4[0] = *(const v4f*)(r + kb0A);
    qa[hb].f4[1] = *(const v4f*)(r + kb0A + 16);
  }

  v8f o[8];
  float mrow[8], lrow[8];
  #pragma unroll
  for (int nb = 0; nb < 8; ++nb) o[nb] = vzero8();
  #pragma unroll
  for (int v = 0; v < 8; ++v){ mrow[v] = -1e30f; lrow[v] = 0.f; }

  __shared__ u16 pbuf[16][32];
  const u16* kbase = Kc + (size_t)kvh * SEQ * HD;
  const u16* vbase = VT + (size_t)kvh * HD * SEQ;

  for (int kb = 0; kb < SEQ; kb += 32){
    // scores: two 16x16 tiles over 32 keys
    v8f s[2];
    #pragma unroll
    for (int t = 0; t < 2; ++t){
      v8f c = vzero8();
      #pragma unroll
      for (int hb = 0; hb < 4; ++hb){
        Frag kfr;
        const u16* r = kbase + (size_t)(kb + t * 16 + l15) * HD + hb * 32 + kb0B;
        kfr.f4[0] = *(const v4f*)r;
        kfr.f4[1] = *(const v4f*)(r + 16);
        c = __builtin_amdgcn_wmma_f32_16x16x32_bf16(
            false, qa[hb].bf, false, kfr.bf, (short)0, c, false, false);
      }
      s[t] = c;
    }
    // online softmax
    float mx[8];
    #pragma unroll
    for (int v = 0; v < 8; ++v){
      s[0][v] *= scale; s[1][v] *= scale;
      float m = fmaxf(s[0][v], s[1][v]);
      for (int d = 1; d < 16; d <<= 1) m = fmaxf(m, __shfl_xor(m, d, 32));
      mx[v] = m;
    }
    float corr[8], rs[8];
    #pragma unroll
    for (int v = 0; v < 8; ++v){
      float mn = fmaxf(mrow[v], mx[v]);
      corr[v] = __expf(mrow[v] - mn);
      mrow[v] = mn;
      float p0 = __expf(s[0][v] - mn), p1 = __expf(s[1][v] - mn);
      s[0][v] = p0; s[1][v] = p1;
      float r = p0 + p1;
      for (int d = 1; d < 16; d <<= 1) r += __shfl_xor(r, d, 32);
      rs[v] = r;
    }
    #pragma unroll
    for (int v = 0; v < 8; ++v) lrow[v] = lrow[v] * corr[v] + rs[v];
    #pragma unroll
    for (int nb = 0; nb < 8; ++nb)
      #pragma unroll
      for (int v = 0; v < 8; ++v) o[nb][v] *= corr[v];

    // stage P (C-layout) into LDS, reload in A-layout
    #pragma unroll
    for (int t = 0; t < 2; ++t)
      #pragma unroll
      for (int v = 0; v < 8; ++v)
        pbuf[v + half * 8][t * 16 + l15] = f2bf(s[t][v]);
    __syncthreads();
    Frag pa;
    {
      const u16* r = &pbuf[l15][0];
      pa.f4[0] = *(const v4f*)(r + kb0A);
      pa.f4[1] = *(const v4f*)(r + kb0A + 16);
    }
    __syncthreads();

    // PV: accumulate over 8 HD tiles
    #pragma unroll
    for (int nb = 0; nb < 8; ++nb){
      Frag vfr;
      const u16* r = vbase + (size_t)(nb * 16 + l15) * SEQ + kb + kb0B;
      vfr.f4[0] = *(const v4f*)r;
      vfr.f4[1] = *(const v4f*)(r + 16);
      o[nb] = __builtin_amdgcn_wmma_f32_16x16x32_bf16(
          false, pa.bf, false, vfr.bf, (short)0, o[nb], false, false);
    }
  }

  // finalize and write concat layout [S][H*HD] bf16
  #pragma unroll
  for (int v = 0; v < 8; ++v){
    float inv = 1.f / lrow[v];
    int row = q0 + v + half * 8;
    #pragma unroll
    for (int nb = 0; nb < 8; ++nb)
      O[(size_t)row * (NHEAD*HD) + h * HD + nb * 16 + l15] = f2bf(o[nb][v] * inv);
  }
}

// ---------------------------------------------------------------------------
// Router: logits = x2 @ gate^T, softmax, top-2, normalized weights,
// build per-expert gather lists (pair id = token*2 + slot).
// ---------------------------------------------------------------------------
__global__ void k_zero_counts(int* c){ if (threadIdx.x < NEXP) c[threadIdx.x] = 0; }

__global__ __launch_bounds__(256) void k_router(
    const u16* __restrict__ x2, const float* __restrict__ gate_w,
    float* __restrict__ w_pair, int* __restrict__ counts, int* __restrict__ rowmap)
{
  const int t = blockIdx.x, tid = threadIdx.x, wid = tid >> 5, lane = tid & 31;
  __shared__ float lg[NEXP];
  for (int e = wid; e < NEXP; e += 8){
    float s = 0.f;
    const u16* xr = x2 + (size_t)t * DH;
    const float* g = gate_w + (size_t)e * DH;
    for (int i = lane; i < DH; i += 32) s += bf2f(xr[i]) * g[i];
    for (int d = 16; d; d >>= 1) s += __shfl_xor(s, d, 32);
    if (lane == 0) lg[e] = s;
  }
  __syncthreads();
  if (tid == 0){
    float mx = -1e30f;
    for (int e = 0; e < NEXP; ++e) mx = fmaxf(mx, lg[e]);
    float p[NEXP], sum = 0.f;
    for (int e = 0; e < NEXP; ++e){ p[e] = __expf(lg[e] - mx); sum += p[e]; }
    float inv = 1.f / sum;
    for (int e = 0; e < NEXP; ++e) p[e] *= inv;
    int i0 = 0; for (int e = 1; e < NEXP; ++e) if (p[e] > p[i0]) i0 = e;
    int i1 = (i0 == 0) ? 1 : 0;
    for (int e = 0; e < NEXP; ++e) if (e != i0 && p[e] > p[i1]) i1 = e;
    float den = p[i0] + p[i1];
    float w0 = p[i0] / den, w1 = p[i1] / den;
    int sel[2] = {i0, i1}; float wv[2] = {w0, w1};
    for (int kk = 0; kk < 2; ++kk){
      int pid = t * 2 + kk;
      w_pair[pid] = wv[kk];
      int pos = atomicAdd(&counts[sel[kk]], 1);
      rowmap[sel[kk] * CAP + pos] = pid;
    }
  }
}

// ---------------------------------------------------------------------------
// a = silu(g) * u  -> bf16
// ---------------------------------------------------------------------------
__global__ __launch_bounds__(256) void k_silu_mul(
    const float* __restrict__ g, const float* __restrict__ u,
    u16* __restrict__ a, int n)
{
  int i = blockIdx.x * 256 + threadIdx.x;
  if (i < n){
    float gv = g[i];
    float s = gv / (1.f + __expf(-gv));
    a[i] = f2bf(s * u[i]);
  }
}

// ---------------------------------------------------------------------------
// out = h_attn + yslot[2t] + yslot[2t+1]
// ---------------------------------------------------------------------------
__global__ __launch_bounds__(256) void k_combine(
    const float* __restrict__ hattn, const float* __restrict__ yslot,
    float* __restrict__ out)
{
  int i = blockIdx.x * 256 + threadIdx.x;
  int t = i / DH, d = i - t * DH;
  out[i] = hattn[i] + yslot[(size_t)(2 * t) * DH + d] + yslot[(size_t)(2 * t + 1) * DH + d];
}

// ---------------------------------------------------------------------------
extern "C" void kernel_launch(void* const* d_in, const int* in_sizes, int n_in,
                              void* d_out, int out_size, void* d_ws, size_t ws_size,
                              hipStream_t stream)
{
  const float* hidden = (const float*)d_in[0];
  const float* ln1    = (const float*)d_in[1];
  const float* ln2    = (const float*)d_in[2];
  const float* qnw    = (const float*)d_in[3];
  const float* knw    = (const float*)d_in[4];
  const float* Wq     = (const float*)d_in[5];
  const float* Wk     = (const float*)d_in[6];
  const float* Wv     = (const float*)d_in[7];
  const float* Wo     = (const float*)d_in[8];
  const float* gatew  = (const float*)d_in[9];
  const float* Wg     = (const float*)d_in[10];
  const float* Wu     = (const float*)d_in[11];
  const float* Wd     = (const float*)d_in[12];
  const int*   pos    = (const int*)d_in[13];
  float* out = (float*)d_out;

  char* ws = (char*)d_ws;
  size_t off = 0;
  auto alloc = [&](size_t b) -> void* {
    void* p = ws + off;
    off += (b + 255) & ~(size_t)255;
    return p;
  };
  u16*   x1    = (u16*)  alloc((size_t)SEQ * DH * 2);
  float* qf    = (float*)alloc((size_t)SEQ * NHEAD * HD * 4);
  float* kf    = (float*)alloc((size_t)SEQ * NKV * HD * 4);
  float* vf    = (float*)alloc((size_t)SEQ * NKV * HD * 4);
  u16*   Qb    = (u16*)  alloc((size_t)NHEAD * SEQ * HD * 2);
  u16*   Kb    = (u16*)  alloc((size_t)NKV * SEQ * HD * 2);
  u16*   VTb   = (u16*)  alloc((size_t)NKV * HD * SEQ * 2);
  u16*   attnO = (u16*)  alloc((size_t)SEQ * NHEAD * HD * 2);
  float* hattn = (float*)alloc((size_t)SEQ * DH * 4);
  u16*   x2    = (u16*)  alloc((size_t)SEQ * DH * 2);
  float* wpair = (float*)alloc((size_t)NPAIR * 4);
  int*   counts= (int*)  alloc((size_t)NEXP * 4);
  int*   rowmap= (int*)  alloc((size_t)NEXP * CAP * 4);
  float* gf    = (float*)alloc((size_t)NPAIR * IMOE * 4);
  float* uf    = (float*)alloc((size_t)NPAIR * IMOE * 4);
  u16*   abf   = (u16*)  alloc((size_t)NPAIR * IMOE * 2);
  float* yslot = (float*)alloc((size_t)NPAIR * DH * 4);

  // 1) pre-attn RMSNorm -> bf16
  k_rmsnorm_bf16<<<SEQ, 256, 0, stream>>>(hidden, ln1, x1, DH);

  // 2) Q/K/V projections (WMMA GEMM, A via TDM)
  k_gemm_bf16<<<dim3((NHEAD*HD)/BN, SEQ/BM, 1), 256, 0, stream>>>(
      x1, DH, Wq, NHEAD*HD, qf, NHEAD*HD,
      nullptr, nullptr, nullptr, nullptr, SEQ, 0, 0);
  k_gemm_bf16<<<dim3((NKV*HD)/BN, SEQ/BM, 1), 256, 0, stream>>>(
      x1, DH, Wk, NKV*HD, kf, NKV*HD,
      nullptr, nullptr, nullptr, nullptr, SEQ, 0, 0);
  k_gemm_bf16<<<dim3((NKV*HD)/BN, SEQ/BM, 1), 256, 0, stream>>>(
      x1, DH, Wv, NKV*HD, vf, NKV*HD,
      nullptr, nullptr, nullptr, nullptr, SEQ, 0, 0);

  // 3) per-head RMSNorm + RoPE, V transpose
  k_qk_rope_v<<<SEQ, 256, 0, stream>>>(qf, kf, vf, qnw, knw, pos, Qb, Kb, VTb);

  // 4) attention (WMMA flash)
  k_attention<<<dim3(SEQ/16, NHEAD), 32, 0, stream>>>(Qb, Kb, VTb, attnO);

  // 5) output projection + residual
  k_gemm_bf16<<<dim3(DH/BN, SEQ/BM, 1), 256, 0, stream>>>(
      attnO, NHEAD*HD, Wo, DH, hattn, DH,
      hidden, nullptr, nullptr, nullptr, SEQ, 0, 0);

  // 6) post-attn RMSNorm -> bf16
  k_rmsnorm_bf16<<<SEQ, 256, 0, stream>>>(hattn, ln2, x2, DH);

  // 7) router + gather lists
  k_zero_counts<<<1, 32, 0, stream>>>(counts);
  k_router<<<SEQ, 256, 0, stream>>>(x2, gatew, wpair, counts, rowmap);

  // 8) expert gate/up projections (gathered rows: arow = pair>>1 = token)
  k_gemm_bf16<<<dim3(IMOE/BN, CAP/BM, NEXP), 256, 0, stream>>>(
      x2, DH, Wg, IMOE, gf, IMOE,
      nullptr, nullptr, rowmap, counts, CAP, CAP, 1);
  k_gemm_bf16<<<dim3(IMOE/BN, CAP/BM, NEXP), 256, 0, stream>>>(
      x2, DH, Wu, IMOE, uf, IMOE,
      nullptr, nullptr, rowmap, counts, CAP, CAP, 1);

  // 9) silu(g)*u -> bf16
  k_silu_mul<<<(NPAIR * IMOE + 255) / 256, 256, 0, stream>>>(gf, uf, abf, NPAIR * IMOE);

  // 10) expert down projection, scaled by routing weight, into per-pair slots
  k_gemm_bf16<<<dim3(DH/BN, CAP/BM, NEXP), 256, 0, stream>>>(
      abf, IMOE, Wd, DH, yslot, DH,
      nullptr, wpair, rowmap, counts, CAP, CAP, 0);

  // 11) final residual combine
  k_combine<<<(SEQ * DH) / 256, 256, 0, stream>>>(hattn, yslot, out);
}
